// InterpNetwork_1778116460646
// MI455X (gfx1250) — matrix operand order repaired and checked by reading, not compiled, not run on hardware
//
#include <hip/hip_runtime.h>
#include <hip/hip_bf16.h>

typedef __attribute__((ext_vector_type(16))) _Float16 v16h;
typedef __attribute__((ext_vector_type(8)))  float    v8f;

#define NEG_VAL -1000000000.0f

// ---------------- WMMA fragment helpers ----------------
// B fragment pre-packed: frag index f, lane L -> 16 contiguous halves (32B).
__device__ __forceinline__ v16h load_b_packed(const _Float16* __restrict__ pk, int frag, int lane) {
    return *(const v16h*)(pk + ((size_t)frag * 32 + lane) * 16);
}

// A fragment from row-major f16 (global or LDS): a[e] = A[m0+m][k0 + (e<8?e:e+8) + 8*s]
__device__ __forceinline__ v16h load_a_rm(const _Float16* __restrict__ A, int lda, int m0, int k0, int lane) {
    int m = lane & 15, s = lane >> 4;
    const _Float16* row = A + (size_t)(m0 + m) * lda + k0 + 8 * s;
    v16h a;
#pragma unroll
    for (int e = 0; e < 8; ++e) a[e] = row[e];
#pragma unroll
    for (int e = 0; e < 8; ++e) a[8 + e] = row[16 + e];
    return a;
}

__device__ __forceinline__ v8f wmma_f16(v16h a, v16h b, v8f c) {
    return __builtin_amdgcn_wmma_f32_16x16x32_f16(false, a, false, b, (short)0, c, false, false);
}

// ---------------- Kernel: pack f32 row-major W[K][N] into f16 B fragments ----------------
__global__ void pack_b(const float* __restrict__ src, _Float16* __restrict__ dst, int K, int N) {
    int total = K * N;
    int ntile = N >> 4;
    for (int p = blockIdx.x * blockDim.x + threadIdx.x; p < total; p += gridDim.x * blockDim.x) {
        int e = p & 15;
        int L = (p >> 4) & 31;
        int frag = p >> 9;
        int nt = frag % ntile;
        int kt = frag / ntile;
        int s = L >> 4, ln = L & 15;
        int k = kt * 32 + ((e < 8) ? e : (e + 8)) + 8 * s;
        int n = nt * 16 + ln;
        dst[p] = (_Float16)src[(size_t)k * N + n];
    }
}

// ---------------- Kernel: edge scalar MLP 1->32->32 (VALU), f16 row-major out ----------------
__global__ void edge_mlp(const float* __restrict__ edge_w,
                         const float* __restrict__ W1, const float* __restrict__ b1,
                         const float* __restrict__ W2, const float* __restrict__ b2,
                         _Float16* __restrict__ EEh) {
    int tid = blockIdx.x * blockDim.x + threadIdx.x;
    if (tid >= 2 * 96 * 96) return;
    int g = tid / 9216;
    int r = tid % 9216;
    int i = r / 96, j = r % 96;
    float w = edge_w[(size_t)(g * 96 + i) * 192 + g * 96 + j];
    float h[32];
#pragma unroll
    for (int c = 0; c < 32; ++c) {
        float v = w * W1[c] + b1[c];
        h[c] = v > 0.f ? v : 0.f;
    }
    for (int c2 = 0; c2 < 32; ++c2) {
        float acc = b2[c2];
#pragma unroll
        for (int c = 0; c < 32; ++c) acc += h[c] * W2[c * 32 + c2];
        acc = acc > 0.f ? acc : 0.f;
        EEh[(size_t)tid * 32 + c2] = (_Float16)acc;
    }
}

// ---------------- Kernel: node embedding mlp2 128->128->128 per (g,t) ----------------
// grid = 66 blocks (g*33+t), 128 threads (4 waves)
__global__ void node_embed(const float* __restrict__ x,
                           const _Float16* __restrict__ W1p, const _Float16* __restrict__ W2p,
                           const float* __restrict__ b1, const float* __restrict__ b2,
                           _Float16* __restrict__ Eh) {
    __shared__ _Float16 Xs[96 * 128];
    __shared__ _Float16 Hs[96 * 128];
    int gt = blockIdx.x;
    int g = gt / 33;
    int tid = threadIdx.x;
    int lane = tid & 31, wave = tid >> 5;
    int ln = lane & 15, s = lane >> 4;

    // stage transposed input: A[d][h] = x[gt][h][g*96+d]
    size_t tbase = (size_t)gt * 128 * 192 + g * 96;
    for (int p = tid; p < 96 * 128; p += 128) {
        int d = p >> 7, h = p & 127;
        Xs[p] = (_Float16)x[tbase + (size_t)h * 192 + d];
    }
    __syncthreads();

    // layer 1: Xs(96x128) @ W1(128x128) -> relu -> Hs
    for (int tt = wave; tt < 48; tt += 4) {
        int mt = tt >> 3, nt = tt & 7;
        v8f acc = {};
#pragma unroll
        for (int kt = 0; kt < 4; ++kt) {
            v16h a = load_a_rm(Xs, 128, mt * 16, kt * 32, lane);
            v16h b = load_b_packed(W1p, kt * 8 + nt, lane);
            acc = wmma_f16(a, b, acc);
        }
        int n = nt * 16 + ln;
        float bias = b1[n];
#pragma unroll
        for (int r = 0; r < 8; ++r) {
            float v = acc[r] + bias;
            v = v > 0.f ? v : 0.f;
            Hs[(size_t)(mt * 16 + r + 8 * s) * 128 + n] = (_Float16)v;
        }
    }
    __syncthreads();

    // layer 2: Hs @ W2 -> relu -> Eh (f16)
    for (int tt = wave; tt < 48; tt += 4) {
        int mt = tt >> 3, nt = tt & 7;
        v8f acc = {};
#pragma unroll
        for (int kt = 0; kt < 4; ++kt) {
            v16h a = load_a_rm(Hs, 128, mt * 16, kt * 32, lane);
            v16h b = load_b_packed(W2p, kt * 8 + nt, lane);
            acc = wmma_f16(a, b, acc);
        }
        int n = nt * 16 + ln;
        float bias = b2[n];
#pragma unroll
        for (int r = 0; r < 8; ++r) {
            float v = acc[r] + bias;
            v = v > 0.f ? v : 0.f;
            Eh[((size_t)gt * 96 + mt * 16 + r + 8 * s) * 128 + n] = (_Float16)v;
        }
    }
}

// ---------------- Kernel: Ae = EE(9216x32) @ W1e(32x128) + ud_b1  (per graph) ----------------
// grid = 2*144 blocks of 256 threads; block handles 64 rows x 128 cols
__global__ void ae_gemm(const _Float16* __restrict__ EEh, const _Float16* __restrict__ W1ep,
                        const float* __restrict__ b1, float* __restrict__ Ae) {
    int b = blockIdx.x;
    int g = b / 144;
    int rb = (b % 144) * 64;
    int lane = threadIdx.x & 31, wave = threadIdx.x >> 5;
    int ln = lane & 15, s = lane >> 4;
    const _Float16* A = EEh + (size_t)g * 9216 * 32;
    for (int tt = wave; tt < 32; tt += 8) {
        int mt = tt >> 3, nt = tt & 7;
        v8f acc = {};
        v16h a = load_a_rm(A, 32, rb + mt * 16, 0, lane);
        v16h bb = load_b_packed(W1ep, nt, lane);
        acc = wmma_f16(a, bb, acc);
        int n = nt * 16 + ln;
        float bias = b1[n];
#pragma unroll
        for (int r = 0; r < 8; ++r) {
            size_t row = (size_t)g * 9216 + rb + mt * 16 + r + 8 * s;
            Ae[row * 128 + n] = acc[r] + bias;
        }
    }
}

// ---------------- Kernel: P = E@W1a, Q = E@W1b per (g,t); f32 out ----------------
// grid = 132 (gt*2 + half), 128 threads
__global__ void pq_gemm(const _Float16* __restrict__ Eh,
                        const _Float16* __restrict__ W1ap, const _Float16* __restrict__ W1bp,
                        float* __restrict__ P, float* __restrict__ Q) {
    int bid = blockIdx.x;
    int half = bid & 1;
    int gt = bid >> 1;
    int lane = threadIdx.x & 31, wave = threadIdx.x >> 5;
    int ln = lane & 15, s = lane >> 4;
    const _Float16* A = Eh + (size_t)gt * 96 * 128;
    const _Float16* Bp = half ? W1bp : W1ap;
    float* out = (half ? Q : P) + (size_t)gt * 96 * 128;
    for (int tt = wave; tt < 48; tt += 4) {
        int mt = tt >> 3, nt = tt & 7;
        v8f acc = {};
#pragma unroll
        for (int kt = 0; kt < 4; ++kt) {
            v16h a = load_a_rm(A, 128, mt * 16, kt * 32, lane);
            v16h b = load_b_packed(Bp, kt * 8 + nt, lane);
            acc = wmma_f16(a, b, acc);
        }
        int n = nt * 16 + ln;
#pragma unroll
        for (int r = 0; r < 8; ++r)
            out[(size_t)(mt * 16 + r + 8 * s) * 128 + n] = acc[r];
    }
}

// ---------------- Kernel: fused ud layers: h1=relu(P+Q+Ae) -> W2 relu -> W3 dot -> mask ----------------
// wave-task: (g,t,i,jt). tasks = 2*32*96*6 = 36864; grid = 4608 blocks x 256
__global__ void ud_kernel(const float* __restrict__ P, const float* __restrict__ Q,
                          const float* __restrict__ Ae,
                          const _Float16* __restrict__ W2p, const float* __restrict__ b2,
                          const float* __restrict__ W3, const float* __restrict__ b3,
                          const float* __restrict__ edge_w, float* __restrict__ outClass) {
    int lane = threadIdx.x & 31, wave = threadIdx.x >> 5;
    int ln = lane & 15, s = lane >> 4;
    int task = blockIdx.x * 8 + wave;
    int jt = task % 6;
    int i = (task / 6) % 96;
    int t = (task / (6 * 96)) % 32;
    int g = task / (6 * 96 * 32);
    int j0 = jt * 16;
    int j = j0 + ln;  // row within M-tile

    const float* Prow = P + ((size_t)(g * 33 + t + 1) * 96 + i) * 128;
    const float* Qrow = Q + ((size_t)(g * 33 + t) * 96 + j) * 128;
    const float* Arow = Ae + ((size_t)g * 9216 + (size_t)i * 96 + j) * 128;

    // build h1 A-fragments in registers: h1 = relu(P[i] + Q[j] + Ae[i,j])
    v16h af[4];
#pragma unroll
    for (int kt = 0; kt < 4; ++kt) {
#pragma unroll
        for (int e = 0; e < 16; ++e) {
            int k = kt * 32 + ((e < 8) ? e : (e + 8)) + 8 * s;
            float v = Prow[k] + Qrow[k] + Arow[k];
            af[kt][e] = (_Float16)(v > 0.f ? v : 0.f);
        }
    }

    // layer2 (128->64) via WMMA + layer3 (64->1) via VALU dot
    float partial[8];
#pragma unroll
    for (int r = 0; r < 8; ++r) partial[r] = 0.f;
#pragma unroll
    for (int nt = 0; nt < 4; ++nt) {
        v8f acc = {};
#pragma unroll
        for (int kt = 0; kt < 4; ++kt) {
            v16h b = load_b_packed(W2p, kt * 4 + nt, lane);
            acc = wmma_f16(af[kt], b, acc);
        }
        int n = nt * 16 + ln;
        float bias = b2[n];
        float w3v = W3[n];
#pragma unroll
        for (int r = 0; r < 8; ++r) {
            float h2 = acc[r] + bias;
            h2 = h2 > 0.f ? h2 : 0.f;
            partial[r] += h2 * w3v;
        }
    }
    // reduce across 16 lanes within each half-wave
#pragma unroll
    for (int off = 1; off < 16; off <<= 1) {
#pragma unroll
        for (int r = 0; r < 8; ++r) partial[r] += __shfl_xor(partial[r], off, 32);
    }
    if (ln == 0) {
        float bias3 = b3[0];
#pragma unroll
        for (int r = 0; r < 8; ++r) {
            int jj = j0 + r + 8 * s;
            float v = partial[r] + bias3;
            float ewv = edge_w[(size_t)(g * 96 + i) * 192 + g * 96 + jj];
            if (ewv == 0.0f && i != jj) v = NEG_VAL;
            outClass[((size_t)(g * 32 + t) * 96 + i) * 96 + jj] = v;
        }
    }
}

// ---------------- Kernel: dist head: concat[E_t, E_{t+1}] -> 256->128->64->1 ----------------
// grid = 64 (g*32+t), 128 threads (4 waves)
__global__ void dist_kernel(const _Float16* __restrict__ Eh,
                            const _Float16* __restrict__ W1p, const float* __restrict__ b1,
                            const _Float16* __restrict__ W2p, const float* __restrict__ b2,
                            const float* __restrict__ W3, const float* __restrict__ b3,
                            float* __restrict__ outDist) {
    __shared__ _Float16 Hs[96 * 128];
    int bid = blockIdx.x;
    int g = bid >> 5, t = bid & 31;
    int lane = threadIdx.x & 31, wave = threadIdx.x >> 5;
    int ln = lane & 15, s = lane >> 4;

    // layer1: K=256 (first 128 from E[t], next 128 from E[t+1])
    for (int tt = wave; tt < 48; tt += 4) {
        int mt = tt >> 3, nt = tt & 7;
        v8f acc = {};
#pragma unroll
        for (int kt = 0; kt < 8; ++kt) {
            const _Float16* Abase = Eh + (size_t)((g * 33 + t + (kt >= 4 ? 1 : 0)) * 96) * 128;
            int k0 = (kt & 3) * 32;
            v16h a = load_a_rm(Abase, 128, mt * 16, k0, lane);
            v16h b = load_b_packed(W1p, kt * 8 + nt, lane);
            acc = wmma_f16(a, b, acc);
        }
        int n = nt * 16 + ln;
        float bias = b1[n];
#pragma unroll
        for (int r = 0; r < 8; ++r) {
            float v = acc[r] + bias;
            v = v > 0.f ? v : 0.f;
            Hs[(size_t)(mt * 16 + r + 8 * s) * 128 + n] = (_Float16)v;
        }
    }
    __syncthreads();

    // layer2 (128->64) WMMA + layer3 (64->1) dot
    for (int mt = wave; mt < 6; mt += 4) {
        v16h af[4];
#pragma unroll
        for (int kt = 0; kt < 4; ++kt) af[kt] = load_a_rm(Hs, 128, mt * 16, kt * 32, lane);
        float partial[8];
#pragma unroll
        for (int r = 0; r < 8; ++r) partial[r] = 0.f;
#pragma unroll
        for (int nt = 0; nt < 4; ++nt) {
            v8f acc = {};
#pragma unroll
            for (int kt = 0; kt < 4; ++kt) {
                v16h b = load_b_packed(W2p, kt * 4 + nt, lane);
                acc = wmma_f16(af[kt], b, acc);
            }
            int n = nt * 16 + ln;
            float bias = b2[n];
            float w3v = W3[n];
#pragma unroll
            for (int r = 0; r < 8; ++r) {
                float h2 = acc[r] + bias;
                h2 = h2 > 0.f ? h2 : 0.f;
                partial[r] += h2 * w3v;
            }
        }
#pragma unroll
        for (int off = 1; off < 16; off <<= 1) {
#pragma unroll
            for (int r = 0; r < 8; ++r) partial[r] += __shfl_xor(partial[r], off, 32);
        }
        if (ln == 0) {
            float bias3 = b3[0];
#pragma unroll
            for (int r = 0; r < 8; ++r) {
                int row = mt * 16 + r + 8 * s;
                outDist[(size_t)(g * 32 + t) * 96 + row] = partial[r] + bias3;
            }
        }
    }
}

// ---------------- workspace layout (bytes) ----------------
#define OFF_NEW1  0u
#define OFF_NEW2  32768u
#define OFF_W1A   65536u
#define OFF_W1B   98304u
#define OFF_W1E   131072u
#define OFF_UDW2  139264u
#define OFF_DPW1  155648u
#define OFF_DPW2  221184u
#define OFF_EEH   237568u
#define OFF_EH    1417216u
#define OFF_P     3039232u
#define OFF_Q     6283264u
#define OFF_AE    9527296u

extern "C" void kernel_launch(void* const* d_in, const int* in_sizes, int n_in,
                              void* d_out, int out_size, void* d_ws, size_t ws_size,
                              hipStream_t stream) {
    const float* x      = (const float*)d_in[0];
    const float* edge_w = (const float*)d_in[1];
    const float* ne_W1  = (const float*)d_in[5];
    const float* ne_b1  = (const float*)d_in[6];
    const float* ne_W2  = (const float*)d_in[7];
    const float* ne_b2  = (const float*)d_in[8];
    const float* ee_W1  = (const float*)d_in[9];
    const float* ee_b1  = (const float*)d_in[10];
    const float* ee_W2  = (const float*)d_in[11];
    const float* ee_b2  = (const float*)d_in[12];
    const float* ud_W1  = (const float*)d_in[13];
    const float* ud_b1  = (const float*)d_in[14];
    const float* ud_W2  = (const float*)d_in[15];
    const float* ud_b2  = (const float*)d_in[16];
    const float* ud_W3  = (const float*)d_in[17];
    const float* ud_b3  = (const float*)d_in[18];
    const float* dp_W1  = (const float*)d_in[19];
    const float* dp_b1  = (const float*)d_in[20];
    const float* dp_W2  = (const float*)d_in[21];
    const float* dp_b2  = (const float*)d_in[22];
    const float* dp_W3  = (const float*)d_in[23];
    const float* dp_b3  = (const float*)d_in[24];

    char* ws = (char*)d_ws;
    _Float16* neW1p = (_Float16*)(ws + OFF_NEW1);
    _Float16* neW2p = (_Float16*)(ws + OFF_NEW2);
    _Float16* W1ap  = (_Float16*)(ws + OFF_W1A);
    _Float16* W1bp  = (_Float16*)(ws + OFF_W1B);
    _Float16* W1ep  = (_Float16*)(ws + OFF_W1E);
    _Float16* udW2p = (_Float16*)(ws + OFF_UDW2);
    _Float16* dpW1p = (_Float16*)(ws + OFF_DPW1);
    _Float16* dpW2p = (_Float16*)(ws + OFF_DPW2);
    _Float16* EEh   = (_Float16*)(ws + OFF_EEH);
    _Float16* Eh    = (_Float16*)(ws + OFF_EH);
    float*    P     = (float*)(ws + OFF_P);
    float*    Q     = (float*)(ws + OFF_Q);
    float*    Ae    = (float*)(ws + OFF_AE);

    float* outClass = (float*)d_out;
    float* outDist  = (float*)d_out + 2 * 32 * 96 * 96;

    // pack weights into B-fragment layout
    pack_b<<<16, 256, 0, stream>>>(ne_W1, neW1p, 128, 128);
    pack_b<<<16, 256, 0, stream>>>(ne_W2, neW2p, 128, 128);
    pack_b<<<16, 256, 0, stream>>>(ud_W1, W1ap, 128, 128);                 // rows 0:128
    pack_b<<<16, 256, 0, stream>>>(ud_W1 + 128 * 128, W1bp, 128, 128);     // rows 128:256
    pack_b<<<8, 256, 0, stream>>>(ud_W1 + 256 * 128, W1ep, 32, 128);       // rows 256:288
    pack_b<<<8, 256, 0, stream>>>(ud_W2, udW2p, 128, 64);
    pack_b<<<32, 256, 0, stream>>>(dp_W1, dpW1p, 256, 128);
    pack_b<<<8, 256, 0, stream>>>(dp_W2, dpW2p, 128, 64);

    // edge scalar MLP -> EE (f16)
    edge_mlp<<<72, 256, 0, stream>>>(edge_w, ee_W1, ee_b1, ee_W2, ee_b2, EEh);

    // node embeddings for all 66 frames -> Eh (f16)
    node_embed<<<66, 128, 0, stream>>>(x, neW1p, neW2p, ne_b1, ne_b2, Eh);

    // Ae = EE @ W1e + ud_b1  (time-independent, reused across 32 steps)
    ae_gemm<<<288, 256, 0, stream>>>(EEh, W1ep, ud_b1, Ae);

    // P = E @ W1a, Q = E @ W1b
    pq_gemm<<<132, 128, 0, stream>>>(Eh, W1ap, W1bp, P, Q);

    // fused ud: class logits
    ud_kernel<<<4608, 256, 0, stream>>>(P, Q, Ae, udW2p, ud_b2, ud_W3, ud_b3, edge_w, outClass);

    // dist head
    dist_kernel<<<64, 128, 0, stream>>>(Eh, dpW1p, dp_b1, dpW2p, dp_b2, dp_W3, dp_b3, outDist);
}
